// MultiHeadAttention_57071525430210
// MI455X (gfx1250) — matrix operand order repaired and checked by reading, not compiled, not run on hardware
//
#include <hip/hip_runtime.h>
#include <hip/hip_bf16.h>

typedef _Float16 h16_t;
typedef _Float16 v16h __attribute__((ext_vector_type(16)));
typedef _Float16 v8h  __attribute__((ext_vector_type(8)));
typedef _Float16 v4h  __attribute__((ext_vector_type(4)));
typedef float    v8f  __attribute__((ext_vector_type(8)));
typedef float    v4f  __attribute__((ext_vector_type(4)));

#define S_LEN 2048
#define D_DIM 1024
#define NHEAD 16
#define DHEAD 64
#define BATCH 2

union FragH { v16h f; v8h h[2]; };

__device__ __forceinline__ v8f wmma_f16(const FragH& a, const FragH& b, v8f c) {
  // D = A(16x32 f16) * B(32x16 f16) + C(16x16 f32)
  return __builtin_amdgcn_wmma_f32_16x16x32_f16(false, a.f, false, b.f,
                                                (short)0, c, false, false);
}

// ---------------------------------------------------------------------------
// 128x128-tile GEMM: out = A @ W^T + bias.
// MODE 0: A fp32, output f16 in head-split [B,H,S,DH] layout (QKV projection)
// MODE 1: A f16,  output fp32 in [M,N] layout (final projection)
// Block: 128 threads = 4 waves in 2x2, each wave computes a 64x64 tile.
// ---------------------------------------------------------------------------
template<int MODE>
__global__ __launch_bounds__(128)
void gemm128(const float* __restrict__ A32, const h16_t* __restrict__ A16,
             const float* __restrict__ W,   const float* __restrict__ bias,
             h16_t* __restrict__ out16,     float* __restrict__ out32)
{
  __shared__ h16_t As[128 * 40];   // [row][k] stride 40 halves (80B, 16B-mult)
  __shared__ h16_t Bs[128 * 40];   // W tile, native [n][k] layout

  const int tid  = threadIdx.x;
  const int lane = tid & 31;
  const int wv   = tid >> 5;
  const int lr   = lane & 15;      // N / M-low index
  const int lg   = lane >> 4;      // lane-group select
  const int m0   = blockIdx.y * 128;
  const int n0   = blockIdx.x * 128;
  const int wm   = (wv >> 1) * 64;
  const int wn   = (wv & 1) * 64;

  v8f acc[4][4];
#pragma unroll
  for (int i = 0; i < 4; ++i)
#pragma unroll
    for (int j = 0; j < 4; ++j) acc[i][j] = (v8f){};

  for (int kk = 0; kk < D_DIM; kk += 32) {
    // ---- stage A tile (128 x 32) into LDS as f16 ----
    if (MODE == 0) {
#pragma unroll
      for (int i = 0; i < 8; ++i) {
        int c = tid + i * 128;                 // 1024 float4 chunks
        int row = c >> 3, c4 = (c & 7) * 4;
        v4f x = *(const v4f*)(A32 + (size_t)(m0 + row) * D_DIM + kk + c4);
        v4h hx = { (h16_t)x.x, (h16_t)x.y, (h16_t)x.z, (h16_t)x.w };
        *(v4h*)&As[row * 40 + c4] = hx;
      }
    } else {
#pragma unroll
      for (int i = 0; i < 4; ++i) {
        int c = tid + i * 128;                 // 512 v8h chunks
        int row = c >> 2, c8 = (c & 3) * 8;
        *(v8h*)&As[row * 40 + c8] =
            *(const v8h*)(A16 + (size_t)(m0 + row) * D_DIM + kk + c8);
      }
    }
    // ---- stage W tile (128 x 32, [n][k] native) ----
#pragma unroll
    for (int i = 0; i < 8; ++i) {
      int c = tid + i * 128;
      int row = c >> 3, c4 = (c & 7) * 4;
      v4f x = *(const v4f*)(W + (size_t)(n0 + row) * D_DIM + kk + c4);
      v4h hx = { (h16_t)x.x, (h16_t)x.y, (h16_t)x.z, (h16_t)x.w };
      *(v4h*)&Bs[row * 40 + c4] = hx;
    }
    __syncthreads();

    // ---- fragments from LDS (b128 loads, 16B aligned) ----
    FragH af[4], bf[4];
#pragma unroll
    for (int i = 0; i < 4; ++i) {
      int base = (wm + i * 16 + lr) * 40 + lg * 8;
      af[i].h[0] = *(const v8h*)&As[base];        // K = 8*lg + 0..7
      af[i].h[1] = *(const v8h*)&As[base + 16];   // K = 16 + 8*lg + 0..7
    }
#pragma unroll
    for (int j = 0; j < 4; ++j) {
      int base = (wn + j * 16 + lr) * 40 + lg * 8;
      bf[j].h[0] = *(const v8h*)&Bs[base];        // K = 8*lg + 0..7
      bf[j].h[1] = *(const v8h*)&Bs[base + 16];   // K = 16 + 8*lg + 0..7
    }
#pragma unroll
    for (int i = 0; i < 4; ++i)
#pragma unroll
      for (int j = 0; j < 4; ++j)
        acc[i][j] = wmma_f16(af[i], bf[j], acc[i][j]);
    __syncthreads();
  }

  // ---- epilogue: C/D layout  M = r + 8*lg, N = lr ----
#pragma unroll
  for (int j = 0; j < 4; ++j) {
    int n = n0 + wn + j * 16 + lr;
    float bn = bias[n];
#pragma unroll
    for (int i = 0; i < 4; ++i) {
#pragma unroll
      for (int r = 0; r < 8; ++r) {
        int m = m0 + wm + i * 16 + r + 8 * lg;
        float val = acc[i][j][r] + bn;
        if (MODE == 0) {
          int bb = m >> 11, s = m & (S_LEN - 1);
          int h  = n >> 6,  dh = n & (DHEAD - 1);
          out16[(((size_t)(bb * NHEAD + h)) * S_LEN + s) * DHEAD + dh] = (h16_t)val;
        } else {
          out32[(size_t)m * D_DIM + n] = val;
        }
      }
    }
  }
}

// ---------------------------------------------------------------------------
// Flash attention. Grid: (S/64 q-blocks, B*H). Block: 128 threads = 4 waves,
// each wave owns 16 query rows. Q/K fragments loaded directly from global f16
// (layout [bh][s][d] is fragment-native); V staged transposed in LDS; P goes
// through per-wave LDS tile to swap C/D layout -> A-fragment layout.
// ---------------------------------------------------------------------------
__global__ __launch_bounds__(128)
void attn_kernel(const h16_t* __restrict__ Qh, const h16_t* __restrict__ Kh,
                 const h16_t* __restrict__ Vh, const int* __restrict__ mask,
                 h16_t* __restrict__ Oout)
{
  __shared__ h16_t Vt[64 * 72];        // V^T tile: [d][s'] stride 72 halves
  __shared__ h16_t Pm[4][16 * 72];     // per-wave P tile: [row][s']

  const int tid  = threadIdx.x;
  const int lane = tid & 31;
  const int wv   = tid >> 5;
  const int lr   = lane & 15;
  const int lg   = lane >> 4;
  const int qb   = blockIdx.x;
  const int bh   = blockIdx.y;
  const int bb   = bh >> 4;            // batch
  const int hh   = bh & 15;            // head
  const int q0   = qb * 64;
  const size_t bhS = (size_t)bh * S_LEN;

  // --- Q A-fragments for this wave's 16 rows (resident whole kernel) ---
  FragH qf[2];
  {
    const h16_t* qp = Qh + (bhS + q0 + wv * 16 + lr) * DHEAD;
#pragma unroll
    for (int ks = 0; ks < 2; ++ks) {
      qf[ks].h[0] = *(const v8h*)(qp + ks * 32 + lg * 8);
      qf[ks].h[1] = *(const v8h*)(qp + ks * 32 + 16 + lg * 8);
    }
  }

  v8f oacc[4];
#pragma unroll
  for (int j = 0; j < 4; ++j) oacc[j] = (v8f){};
  float mrow[8], lrow[8];
#pragma unroll
  for (int r = 0; r < 8; ++r) { mrow[r] = -3.0e38f; lrow[r] = 0.f; }

  for (int kv = 0; kv < S_LEN; kv += 64) {
    __syncthreads();                   // protect Vt against prior readers
    // ---- stage V^T (64x64 halves) ----
#pragma unroll
    for (int i = 0; i < 4; ++i) {
      int c = tid + i * 128;           // 512 v8h chunks
      int sp = c >> 3, c8 = (c & 7) * 8;
      v8h vv = *(const v8h*)(Vh + (bhS + kv + sp) * DHEAD + c8);
#pragma unroll
      for (int e = 0; e < 8; ++e)
        Vt[(c8 + e) * 72 + sp] = vv[e];
    }
    __syncthreads();

    // ---- scores = Q @ K^T  (B-frags straight from global f16) ----
    v8f sc[4];
#pragma unroll
    for (int t = 0; t < 4; ++t) {
      const h16_t* kp = Kh + (bhS + kv + t * 16 + lr) * DHEAD;
      FragH kf0, kf1;
      kf0.h[0] = *(const v8h*)(kp + lg * 8);
      kf0.h[1] = *(const v8h*)(kp + lg * 8 + 16);
      kf1.h[0] = *(const v8h*)(kp + 32 + lg * 8);
      kf1.h[1] = *(const v8h*)(kp + 32 + lg * 8 + 16);
      sc[t] = wmma_f16(qf[0], kf0, (v8f){});
      sc[t] = wmma_f16(qf[1], kf1, sc[t]);
    }

    // ---- mask + online softmax (rows live in 16-lane groups) ----
    const int qrow_base = q0 + wv * 16 + 8 * lg;
#pragma unroll
    for (int r = 0; r < 8; ++r) {
      const int qi = qrow_base + r;
      const int* mp = mask + (size_t)qi * S_LEN + kv + lr;
      float sv[4];
      float tmax = -3.0e38f;
#pragma unroll
      for (int t = 0; t < 4; ++t) {
        int mv = mp[t * 16];
        float x = sc[t][r] * 0.125f;           // 1/sqrt(64)
        x = mv ? x : -1.0e9f;
        sv[t] = x;
        tmax = fmaxf(tmax, x);
      }
#pragma unroll
      for (int xm = 1; xm <= 8; xm <<= 1)
        tmax = fmaxf(tmax, __shfl_xor(tmax, xm, 32));
      float mnew = fmaxf(mrow[r], tmax);
      float corr = __expf(mrow[r] - mnew);
      mrow[r] = mnew;
      float rsum = 0.f;
#pragma unroll
      for (int t = 0; t < 4; ++t) {
        float p = __expf(sv[t] - mnew);
        sc[t][r] = p;
        rsum += p;
      }
#pragma unroll
      for (int xm = 1; xm <= 8; xm <<= 1)
        rsum += __shfl_xor(rsum, xm, 32);
      lrow[r] = lrow[r] * corr + rsum;
#pragma unroll
      for (int j = 0; j < 4; ++j)
        oacc[j][r] *= corr;
    }

    // ---- P (C/D layout) -> per-wave LDS tile as f16 ----
#pragma unroll
    for (int t = 0; t < 4; ++t)
#pragma unroll
      for (int r = 0; r < 8; ++r)
        Pm[wv][(r + 8 * lg) * 72 + t * 16 + lr] = (h16_t)sc[t][r];

    // ---- O += P @ V ----
#pragma unroll
    for (int ks = 0; ks < 2; ++ks) {
      FragH pf;
      int pb = lr * 72 + ks * 32 + lg * 8;
      pf.h[0] = *(const v8h*)&Pm[wv][pb];
      pf.h[1] = *(const v8h*)&Pm[wv][pb + 16];
#pragma unroll
      for (int j = 0; j < 4; ++j) {
        FragH vf;
        int vb = (j * 16 + lr) * 72 + ks * 32 + lg * 8;
        vf.h[0] = *(const v8h*)&Vt[vb];
        vf.h[1] = *(const v8h*)&Vt[vb + 16];
        oacc[j] = wmma_f16(pf, vf, oacc[j]);
      }
    }
  }

  // ---- normalize and write O as f16 in [B,S,D] layout ----
#pragma unroll
  for (int r = 0; r < 8; ++r) {
    int si = q0 + wv * 16 + r + 8 * lg;
    float inv = 1.0f / lrow[r];
    size_t base = ((size_t)bb * S_LEN + si) * D_DIM + hh * DHEAD;
#pragma unroll
    for (int j = 0; j < 4; ++j)
      Oout[base + j * 16 + lr] = (h16_t)(oacc[j][r] * inv);
  }
}

// ---------------------------------------------------------------------------
extern "C" void kernel_launch(void* const* d_in, const int* in_sizes, int n_in,
                              void* d_out, int out_size, void* d_ws, size_t ws_size,
                              hipStream_t stream)
{
  const float* q    = (const float*)d_in[0];
  const float* k    = (const float*)d_in[1];
  const float* v    = (const float*)d_in[2];
  const int*   mask = (const int*)  d_in[3];
  const float* Wq   = (const float*)d_in[4];
  const float* bq   = (const float*)d_in[5];
  const float* Wk   = (const float*)d_in[6];
  const float* bk   = (const float*)d_in[7];
  const float* Wv   = (const float*)d_in[8];
  const float* bv   = (const float*)d_in[9];
  const float* Wo   = (const float*)d_in[10];
  const float* bo   = (const float*)d_in[11];
  float* out = (float*)d_out;

  h16_t* ws = (h16_t*)d_ws;
  const size_t N1 = (size_t)BATCH * S_LEN * D_DIM;   // 4,194,304 halves
  h16_t* Qh = ws;
  h16_t* Kh = ws + N1;
  h16_t* Vh = ws + 2 * N1;
  h16_t* Oh = ws + 3 * N1;

  dim3 grid(D_DIM / 128, (BATCH * S_LEN) / 128);     // (8, 32)
  gemm128<0><<<grid, 128, 0, stream>>>(q, nullptr, Wq, bq, Qh, nullptr);
  gemm128<0><<<grid, 128, 0, stream>>>(k, nullptr, Wk, bk, Kh, nullptr);
  gemm128<0><<<grid, 128, 0, stream>>>(v, nullptr, Wv, bv, Vh, nullptr);

  attn_kernel<<<dim3(S_LEN / 64, BATCH * NHEAD), 128, 0, stream>>>(Qh, Kh, Vh, mask, Oh);

  gemm128<1><<<grid, 128, 0, stream>>>(nullptr, Oh, Wo, bo, nullptr, out);
}
